// BilinearDense_35716948033693
// MI455X (gfx1250) — compile-verified, matrix operand order checked by reading, not run on hardware
//
#include <hip/hip_runtime.h>

typedef float        v2f   __attribute__((ext_vector_type(2)));
typedef float        v4f   __attribute__((ext_vector_type(4)));
typedef float        v8f   __attribute__((ext_vector_type(8)));
typedef __bf16       v16bf __attribute__((ext_vector_type(16)));
typedef __bf16       v4bf  __attribute__((ext_vector_type(4)));
typedef unsigned int v4u   __attribute__((ext_vector_type(4)));

#define BTOT 4096
#define D    256           // X_DIM == Z_DIM == OUT_DIM
#define KW   (D * D)       // 65536 = W_kernel row length
#define RS   260           // LDS row stride (floats): 16B-aligned rows, conflict-free
#define SPW  32            // samples per workgroup (2 WMMA M-tiles)

// ---------------------------------------------------------------------------
// One-shot split of W_kernel into bf16 hi/lo planes (hi = rne(w), lo = rne(w - hi))
// ---------------------------------------------------------------------------
__global__ __launch_bounds__(256) void wk_split_bf16(
    const float* __restrict__ Wk,
    unsigned short* __restrict__ hiU, unsigned short* __restrict__ loU)
{
    v4bf* hi = (v4bf*)hiU;
    v4bf* lo = (v4bf*)loU;
    const int total = (D * KW) / 4;
    for (int q = blockIdx.x * blockDim.x + threadIdx.x; q < total;
         q += gridDim.x * blockDim.x) {
        v4f w = ((const v4f*)Wk)[q];
        v4bf h, l;
        #pragma unroll
        for (int j = 0; j < 4; ++j) {
            __bf16 hh = (__bf16)w[j];
            h[j] = hh;
            l[j] = (__bf16)(w[j] - (float)hh);
        }
        hi[q] = h;
        lo[q] = l;
    }
}

// ---------------------------------------------------------------------------
// Fragment helpers (bf16 16x16x32 operand layout, wave32)
// A 16x32: lanes 0-15 hold K {0..7,16..23}, lanes 16-31 hold K {8..15,24..31},
// K-pairs packed per VGPR.  B 32x16 mirrors A with N = lane%16.
// ---------------------------------------------------------------------------
union BFrag { v4u u[2]; v16bf v; };

__device__ __forceinline__ v16bf load_b_frag(const __bf16* base, int i0, int hi) {
    BFrag f;
    f.u[0] = *(const v4u*)(base + i0 + 8 * hi);        // K = 8hi .. 8hi+7
    f.u[1] = *(const v4u*)(base + i0 + 16 + 8 * hi);   // K = 16+8hi .. +7
    return f.v;
}

__device__ __forceinline__ void make_a_frags(const float* xrow, int i0, int hi,
                                             float zv, v16bf& ah, v16bf& al) {
    v4f q0 = *(const v4f*)(xrow + i0 + 8 * hi);        // ds_load_b128
    v4f q1 = *(const v4f*)(xrow + i0 + 8 * hi + 4);
    v4f q2 = *(const v4f*)(xrow + i0 + 16 + 8 * hi);
    v4f q3 = *(const v4f*)(xrow + i0 + 20 + 8 * hi);
    float xf[16] = { q0.x, q0.y, q0.z, q0.w, q1.x, q1.y, q1.z, q1.w,
                     q2.x, q2.y, q2.z, q2.w, q3.x, q3.y, q3.z, q3.w };
    #pragma unroll
    for (int j = 0; j < 16; ++j) {
        float a   = zv * xf[j];        // rank-1 A element in f32
        __bf16 h  = (__bf16)a;         // RNE -> v_cvt_pk_bf16_f32
        ah[j] = h;
        al[j] = (__bf16)(a - (float)h);
    }
}

#define WMMA_BF16(ACC, A, B) \
    ACC = __builtin_amdgcn_wmma_f32_16x16x32_bf16(false, A, false, B, (short)0, ACC, false, false)
#define WMMA_F32(ACC, A, B) \
    ACC = __builtin_amdgcn_wmma_f32_16x16x4_f32(false, A, false, B, (short)0, ACC, false, false)

// ---------------------------------------------------------------------------
// Main fused kernel: bf16x3 split GEMM  out[b,o] = sum_{zk,i} z*x*Wk + biases
// ---------------------------------------------------------------------------
__global__ __launch_bounds__(256) void bilinear_wmma_bf16x3(
    const float* __restrict__ x,  const float* __restrict__ z,
    const unsigned short* __restrict__ WhiU, const unsigned short* __restrict__ WloU,
    const float* __restrict__ Wb, const float* __restrict__ bk,
    const float* __restrict__ bb, float* __restrict__ out)
{
    __shared__ float zs[SPW * RS];
    __shared__ float xs[SPW * RS];

    const int tid   = threadIdx.x;
    const int bbase = blockIdx.x * SPW;

    for (int idx = tid; idx < SPW * D; idx += 256) {
        const int s = idx >> 8, c = idx & (D - 1);
        zs[s * RS + c] = z[(bbase + s) * D + c];
        xs[s * RS + c] = x[(bbase + s) * D + c];
    }
    __syncthreads();

    const int lane = tid & 31;
    const int wave = tid >> 5;
    const int r    = lane & 15;
    const int hi   = lane >> 4;
    const int o0   = (wave * 2 + 0) * 16 + r;
    const int o1   = (wave * 2 + 1) * 16 + r;

    const __bf16* Whi = (const __bf16*)WhiU;
    const __bf16* Wlo = (const __bf16*)WloU;

    v8f acc00 = {}; v8f acc01 = {}; v8f acc10 = {}; v8f acc11 = {};

    const float* xrowA = &xs[(r +  0) * RS];
    const float* xrowB = &xs[(r + 16) * RS];

    // ---- main contraction, bf16 hi/lo split, fp32 accumulate ----
    for (int zk = 0; zk < D; ++zk) {
        const float zvA = zs[(r +  0) * RS + zk];
        const float zvB = zs[(r + 16) * RS + zk];
        const __bf16* bh0 = Whi + (size_t)zk * KW + (size_t)o0 * D;
        const __bf16* bh1 = Whi + (size_t)zk * KW + (size_t)o1 * D;
        const __bf16* bl0 = Wlo + (size_t)zk * KW + (size_t)o0 * D;
        const __bf16* bl1 = Wlo + (size_t)zk * KW + (size_t)o1 * D;
        for (int i0 = 0; i0 < D; i0 += 32) {
            v16bf aAh, aAl, aBh, aBl;
            make_a_frags(xrowA, i0, hi, zvA, aAh, aAl);
            make_a_frags(xrowB, i0, hi, zvB, aBh, aBl);
            const v16bf Bh0 = load_b_frag(bh0, i0, hi);   // 2x global_load_b128
            const v16bf Bl0 = load_b_frag(bl0, i0, hi);
            const v16bf Bh1 = load_b_frag(bh1, i0, hi);
            const v16bf Bl1 = load_b_frag(bl1, i0, hi);
            WMMA_BF16(acc00, aAh, Bh0); WMMA_BF16(acc00, aAh, Bl0); WMMA_BF16(acc00, aAl, Bh0);
            WMMA_BF16(acc01, aAh, Bh1); WMMA_BF16(acc01, aAh, Bl1); WMMA_BF16(acc01, aAl, Bh1);
            WMMA_BF16(acc10, aBh, Bh0); WMMA_BF16(acc10, aBh, Bl0); WMMA_BF16(acc10, aBl, Bh0);
            WMMA_BF16(acc11, aBh, Bh1); WMMA_BF16(acc11, aBh, Bl1); WMMA_BF16(acc11, aBl, Bh1);
        }
    }

    // ---- W_bias pass (exact fp32 WMMA): out += sum_i W_bias[o,i]*x[b,i] ----
    {
        const float* xpA = xrowA + 2 * hi;
        const float* xpB = xrowB + 2 * hi;
        const float* w0  = Wb + o0 * D + 2 * hi;
        const float* w1  = Wb + o1 * D + 2 * hi;
        #pragma unroll 4
        for (int i0 = 0; i0 < D; i0 += 4) {
            const v2f aA  = *(const v2f*)(xpA + i0);
            const v2f aB  = *(const v2f*)(xpB + i0);
            const v2f bf0 = *(const v2f*)(w0 + i0);
            const v2f bf1 = *(const v2f*)(w1 + i0);
            WMMA_F32(acc00, aA, bf0); WMMA_F32(acc01, aA, bf1);
            WMMA_F32(acc10, aB, bf0); WMMA_F32(acc11, aB, bf1);
        }
    }

    // ---- per-sample bias (exact fp32 WMMA): out += sum_zk z[b,zk]*b_kernel[zk,o] ----
    for (int zk = 0; zk < D; zk += 4) {
        v2f aA; aA.x = zs[(r +  0) * RS + zk + 2 * hi];
        /*   */ aA.y = zs[(r +  0) * RS + zk + 2 * hi + 1];
        v2f aB; aB.x = zs[(r + 16) * RS + zk + 2 * hi];
        /*   */ aB.y = zs[(r + 16) * RS + zk + 2 * hi + 1];
        v2f bf0; bf0.x = bk[(zk + 2 * hi) * D + o0];
        /*    */ bf0.y = bk[(zk + 2 * hi + 1) * D + o0];
        v2f bf1; bf1.x = bk[(zk + 2 * hi) * D + o1];
        /*    */ bf1.y = bk[(zk + 2 * hi + 1) * D + o1];
        WMMA_F32(acc00, aA, bf0); WMMA_F32(acc01, aA, bf1);
        WMMA_F32(acc10, aB, bf0); WMMA_F32(acc11, aB, bf1);
    }

    const float bias0 = bb[o0];
    const float bias1 = bb[o1];
    #pragma unroll
    for (int j = 0; j < 8; ++j) {
        const int rowA = bbase +      j + 8 * hi;
        const int rowB = bbase + 16 + j + 8 * hi;
        out[rowA * D + o0] = acc00[j] + bias0;
        out[rowA * D + o1] = acc01[j] + bias1;
        out[rowB * D + o0] = acc10[j] + bias0;
        out[rowB * D + o1] = acc11[j] + bias1;
    }
}

// ---------------------------------------------------------------------------
// Fallback: all-fp32 WMMA fused kernel (used when d_ws can't hold bf16 planes)
// ---------------------------------------------------------------------------
__global__ __launch_bounds__(256) void bilinear_wmma_f32(
    const float* __restrict__ x,  const float* __restrict__ z,
    const float* __restrict__ Wk, const float* __restrict__ Wb,
    const float* __restrict__ bk, const float* __restrict__ bb,
    float* __restrict__ out)
{
    __shared__ float zs[SPW * RS];
    __shared__ float xs[SPW * RS];

    const int tid   = threadIdx.x;
    const int bbase = blockIdx.x * SPW;

    for (int idx = tid; idx < SPW * D; idx += 256) {
        const int s = idx >> 8, c = idx & (D - 1);
        zs[s * RS + c] = z[(bbase + s) * D + c];
        xs[s * RS + c] = x[(bbase + s) * D + c];
    }
    __syncthreads();

    const int lane = tid & 31;
    const int wave = tid >> 5;
    const int r    = lane & 15;
    const int hi   = lane >> 4;
    const int o0   = (wave * 2 + 0) * 16 + r;
    const int o1   = (wave * 2 + 1) * 16 + r;

    v8f acc00 = {}; v8f acc01 = {}; v8f acc10 = {}; v8f acc11 = {};

    const float* xpA = &xs[(r +  0) * RS + 2 * hi];
    const float* xpB = &xs[(r + 16) * RS + 2 * hi];

    for (int zk = 0; zk < D; ++zk) {
        const float zvA = zs[(r +  0) * RS + zk];
        const float zvB = zs[(r + 16) * RS + zk];
        const float* w0 = Wk + (size_t)zk * KW + o0 * D + 2 * hi;
        const float* w1 = Wk + (size_t)zk * KW + o1 * D + 2 * hi;
        #pragma unroll 4
        for (int i0 = 0; i0 < D; i0 += 4) {
            const v2f xA  = *(const v2f*)(xpA + i0);
            const v2f xB  = *(const v2f*)(xpB + i0);
            const v2f bf0 = *(const v2f*)(w0 + i0);
            const v2f bf1 = *(const v2f*)(w1 + i0);
            v2f aA; aA.x = zvA * xA.x; aA.y = zvA * xA.y;
            v2f aB; aB.x = zvB * xB.x; aB.y = zvB * xB.y;
            WMMA_F32(acc00, aA, bf0); WMMA_F32(acc01, aA, bf1);
            WMMA_F32(acc10, aB, bf0); WMMA_F32(acc11, aB, bf1);
        }
    }
    {
        const float* w0 = Wb + o0 * D + 2 * hi;
        const float* w1 = Wb + o1 * D + 2 * hi;
        #pragma unroll 4
        for (int i0 = 0; i0 < D; i0 += 4) {
            const v2f aA  = *(const v2f*)(xpA + i0);
            const v2f aB  = *(const v2f*)(xpB + i0);
            const v2f bf0 = *(const v2f*)(w0 + i0);
            const v2f bf1 = *(const v2f*)(w1 + i0);
            WMMA_F32(acc00, aA, bf0); WMMA_F32(acc01, aA, bf1);
            WMMA_F32(acc10, aB, bf0); WMMA_F32(acc11, aB, bf1);
        }
    }
    for (int zk = 0; zk < D; zk += 4) {
        v2f aA; aA.x = zs[(r +  0) * RS + zk + 2 * hi];
        /*   */ aA.y = zs[(r +  0) * RS + zk + 2 * hi + 1];
        v2f aB; aB.x = zs[(r + 16) * RS + zk + 2 * hi];
        /*   */ aB.y = zs[(r + 16) * RS + zk + 2 * hi + 1];
        v2f bf0; bf0.x = bk[(zk + 2 * hi) * D + o0];
        /*    */ bf0.y = bk[(zk + 2 * hi + 1) * D + o0];
        v2f bf1; bf1.x = bk[(zk + 2 * hi) * D + o1];
        /*    */ bf1.y = bk[(zk + 2 * hi + 1) * D + o1];
        WMMA_F32(acc00, aA, bf0); WMMA_F32(acc01, aA, bf1);
        WMMA_F32(acc10, aB, bf0); WMMA_F32(acc11, aB, bf1);
    }

    const float bias0 = bb[o0];
    const float bias1 = bb[o1];
    #pragma unroll
    for (int j = 0; j < 8; ++j) {
        const int rowA = bbase +      j + 8 * hi;
        const int rowB = bbase + 16 + j + 8 * hi;
        out[rowA * D + o0] = acc00[j] + bias0;
        out[rowA * D + o1] = acc01[j] + bias1;
        out[rowB * D + o0] = acc10[j] + bias0;
        out[rowB * D + o1] = acc11[j] + bias1;
    }
}

extern "C" void kernel_launch(void* const* d_in, const int* in_sizes, int n_in,
                              void* d_out, int out_size, void* d_ws, size_t ws_size,
                              hipStream_t stream) {
    const float* x  = (const float*)d_in[0];   // [4096,256]
    const float* z  = (const float*)d_in[1];   // [4096,256]
    const float* Wk = (const float*)d_in[2];   // [256,65536]
    const float* Wb = (const float*)d_in[3];   // [65536]
    const float* bk = (const float*)d_in[4];   // [256,256]
    const float* bb = (const float*)d_in[5];   // [256]
    float* out = (float*)d_out;                // [4096,256]

    const size_t need = (size_t)2 * D * KW * sizeof(unsigned short);  // 67 MB hi+lo planes
    if (ws_size >= need) {
        unsigned short* Whi = (unsigned short*)d_ws;
        unsigned short* Wlo = Whi + (size_t)D * KW;
        wk_split_bf16<<<2048, 256, 0, stream>>>(Wk, Whi, Wlo);
        bilinear_wmma_bf16x3<<<BTOT / SPW, 256, 0, stream>>>(x, z, Whi, Wlo, Wb, bk, bb, out);
    } else {
        bilinear_wmma_f32<<<BTOT / SPW, 256, 0, stream>>>(x, z, Wk, Wb, bk, bb, out);
    }
}
